// KVCacheAttention_1030792151578
// MI455X (gfx1250) — matrix operand order repaired and checked by reading, not compiled, not run on hardware
//
#include <hip/hip_runtime.h>
#include <cstddef>

#define B_   32
#define H_   16
#define DH_  64
#define D_   1024
#define TCTX 4096

typedef __attribute__((ext_vector_type(16))) _Float16 v16h;
typedef __attribute__((ext_vector_type(8)))  float    v8f;

// ---------------------------------------------------------------------------
// Kernel A: fused QKV projection.  q|k|v[b][e] = sum_d x[b][d]*W[d*1024+e] + b[e]
// grid (12 e-chunks of 256, 4 batch-groups of 8), block 256.
// ---------------------------------------------------------------------------
__global__ __launch_bounds__(256)
void qkv_proj_kernel(const float* __restrict__ x,
                     const float* __restrict__ Wq, const float* __restrict__ bq,
                     const float* __restrict__ Wk, const float* __restrict__ bk,
                     const float* __restrict__ Wv, const float* __restrict__ bv,
                     float* __restrict__ qws, float* __restrict__ kws,
                     float* __restrict__ vws)
{
    __shared__ float xs[8 * D_];                       // 32 KB
    const int tid = threadIdx.x;
    const int bg  = blockIdx.y;
    for (int i = tid; i < 8 * D_; i += 256)
        xs[i] = x[(bg * 8 + (i >> 10)) * D_ + (i & (D_ - 1))];
    __syncthreads();

    const int e     = blockIdx.x * 256 + tid;          // 0..3071
    const int which = e >> 10;                         // 0=q 1=k 2=v
    const int ec    = e & (D_ - 1);
    const float* W    = (which == 0) ? Wq : (which == 1) ? Wk : Wv;
    const float* bias = (which == 0) ? bq : (which == 1) ? bk : bv;
    float*       dst  = (which == 0) ? qws : (which == 1) ? kws : vws;

    float acc[8];
#pragma unroll
    for (int j = 0; j < 8; ++j) acc[j] = bias[ec];
    for (int d = 0; d < D_; ++d) {
        const float w = W[d * D_ + ec];
#pragma unroll
        for (int j = 0; j < 8; ++j) acc[j] += xs[j * D_ + d] * w;
    }
#pragma unroll
    for (int j = 0; j < 8; ++j)
        dst[(size_t)(bg * 8 + j) * D_ + ec] = acc[j];
}

// ---------------------------------------------------------------------------
// Kernel B: flash-decode attention.  One block per (b,h); 8 waves x 2 lane-
// halves = 16 independent online-softmax partials (each half owns 8 keys per
// 16-key WMMA tile and all 64 V dims at 4 dims/lane).  Zero cross-lane ops in
// the hot loop; LDS merge at the end.  Wave 0 / half 0 folds in the new token.
// ---------------------------------------------------------------------------
__global__ __launch_bounds__(256)
void decode_attn_kernel(const float* __restrict__ qws,
                        const float* __restrict__ kws,
                        const float* __restrict__ vws,
                        const float* __restrict__ extk,
                        const float* __restrict__ extv,
                        float* __restrict__ ctxws)
{
    const int b    = blockIdx.x >> 4;
    const int h    = blockIdx.x & 15;
    const int tid  = threadIdx.x;
    const int wave = tid >> 5;
    const int lane = tid & 31;
    const int i16  = lane & 15;       // A-tile key row; also dim group (4 dims)
    const int g    = lane >> 4;       // lane half: owns keys g*8..g*8+7
    const int db   = g * 8;

    const float* qv = qws + (size_t)b * D_ + h * DH_;

    // B operand: q (scaled by 1/sqrt(64)) replicated across all 16 columns.
    // 16-bit B 32x16 layout: lanes 0-15 hold K=0..7,16..23, lanes 16-31 K=8..15,24..31.
    v16h qb0, qb1;
#pragma unroll
    for (int i = 0; i < 16; ++i) {
        qb0[i] = (_Float16)(0.125f * qv[ 0 + g * 8 + (i & 7) + (i >> 3) * 16]);
        qb1[i] = (_Float16)(0.125f * qv[32 + g * 8 + (i & 7) + (i >> 3) * 16]);
    }

    const float* Kb = extk + (size_t)b * TCTX * D_ + h * DH_;
    const float* Vb = extv + (size_t)b * TCTX * D_ + h * DH_;

    float m = -1e30f, lsum = 0.f;
    float ax = 0.f, ay = 0.f, az = 0.f, aw = 0.f;
    const int tbeg = wave * 512;

    for (int it = 0; it < 32; ++it) {
        const int t0 = tbeg + it * 16;
        const float* krow = Kb + (size_t)(t0 + i16) * D_;
        // This lane's 32 floats of the 16x64 K tile in 16-bit A-matrix layout:
        // lanes 0-15 take dims {db..db+7, db+16..db+23}, lanes 16-31 (db=8)
        // the complementary octets.
        const float4 f0 = *(const float4*)(krow + db +  0);
        const float4 f1 = *(const float4*)(krow + db +  4);
        const float4 f2 = *(const float4*)(krow + db + 16);
        const float4 f3 = *(const float4*)(krow + db + 20);
        const float4 f4 = *(const float4*)(krow + db + 32);
        const float4 f5 = *(const float4*)(krow + db + 36);
        const float4 f6 = *(const float4*)(krow + db + 48);
        const float4 f7 = *(const float4*)(krow + db + 52);

        if (it + 1 < 32) {            // prefetch next tile (global_prefetch_b8)
            const float* nk = Kb + (size_t)(t0 + 16 + i16) * D_ + db;
            __builtin_prefetch(nk, 0, 0);
            __builtin_prefetch(nk + 32, 0, 0);
            __builtin_prefetch(Vb + (size_t)(t0 + 16 + db) * D_ + 4 * i16, 0, 0);
        }

        v16h a0, a1;
        a0[0]=(_Float16)f0.x; a0[1]=(_Float16)f0.y; a0[2]=(_Float16)f0.z; a0[3]=(_Float16)f0.w;
        a0[4]=(_Float16)f1.x; a0[5]=(_Float16)f1.y; a0[6]=(_Float16)f1.z; a0[7]=(_Float16)f1.w;
        a0[8]=(_Float16)f2.x; a0[9]=(_Float16)f2.y; a0[10]=(_Float16)f2.z; a0[11]=(_Float16)f2.w;
        a0[12]=(_Float16)f3.x; a0[13]=(_Float16)f3.y; a0[14]=(_Float16)f3.z; a0[15]=(_Float16)f3.w;
        a1[0]=(_Float16)f4.x; a1[1]=(_Float16)f4.y; a1[2]=(_Float16)f4.z; a1[3]=(_Float16)f4.w;
        a1[4]=(_Float16)f5.x; a1[5]=(_Float16)f5.y; a1[6]=(_Float16)f5.z; a1[7]=(_Float16)f5.w;
        a1[8]=(_Float16)f6.x; a1[9]=(_Float16)f6.y; a1[10]=(_Float16)f6.z; a1[11]=(_Float16)f6.w;
        a1[12]=(_Float16)f7.x; a1[13]=(_Float16)f7.y; a1[14]=(_Float16)f7.z; a1[15]=(_Float16)f7.w;

        v8f c = {};
        c = __builtin_amdgcn_wmma_f32_16x16x32_f16(false, a0, false, qb0,
                                                   (short)0, c, false, false);
        c = __builtin_amdgcn_wmma_f32_16x16x32_f16(false, a1, false, qb1,
                                                   (short)0, c, false, false);
        // C layout: VGPR j, lanes 0-15 -> key t0+j ; lanes 16-31 -> key t0+8+j.
        // Each half runs its own online softmax over its 8 keys.
        float tmax = c[0];
#pragma unroll
        for (int j = 1; j < 8; ++j) tmax = fmaxf(tmax, c[j]);
        const float mn   = fmaxf(m, tmax);
        const float corr = __expf(m - mn);
        float p[8];
#pragma unroll
        for (int j = 0; j < 8; ++j) p[j] = __expf(c[j] - mn);
        ax *= corr; ay *= corr; az *= corr; aw *= corr;
        lsum *= corr;
        m = mn;

        float tl = 0.f;
#pragma unroll
        for (int j = 0; j < 8; ++j) {
            const float4 vv =
                *(const float4*)(Vb + (size_t)(t0 + db + j) * D_ + 4 * i16);
            tl += p[j];
            ax += p[j] * vv.x;
            ay += p[j] * vv.y;
            az += p[j] * vv.z;
            aw += p[j] * vv.w;
        }
        lsum += tl;
    }

    // Fold in the freshly projected token (position T_CTX): wave 0, half 0.
    if (wave == 0) {
        const float* kn = kws + (size_t)b * D_ + h * DH_;
        const float* vn = vws + (size_t)b * D_ + h * DH_;
        const float4 kk = *(const float4*)(kn + 4 * i16);
        const float4 qq = *(const float4*)(qv + 4 * i16);
        float part = 0.125f * (qq.x * kk.x + qq.y * kk.y +
                               qq.z * kk.z + qq.w * kk.w);
        part += __shfl_xor(part, 1, 32);   // reduce within each 16-lane half
        part += __shfl_xor(part, 2, 32);
        part += __shfl_xor(part, 4, 32);
        part += __shfl_xor(part, 8, 32);
        if (g == 0) {
            const float s    = part;
            const float mn   = fmaxf(m, s);
            const float corr = __expf(m - mn);
            const float ps   = __expf(s - mn);
            const float4 vv  = *(const float4*)(vn + 4 * i16);
            ax = ax * corr + ps * vv.x;
            ay = ay * corr + ps * vv.y;
            az = az * corr + ps * vv.z;
            aw = aw * corr + ps * vv.w;
            lsum = lsum * corr + ps;
            m = mn;
        }
    }

    // Merge the 16 per-half partials through LDS (cold path).
    __shared__ float sm[16], sl[16], sacc[16][DH_];
    const int part = wave * 2 + g;
    if (i16 == 0) { sm[part] = m; sl[part] = lsum; }
    sacc[part][4 * i16 + 0] = ax;
    sacc[part][4 * i16 + 1] = ay;
    sacc[part][4 * i16 + 2] = az;
    sacc[part][4 * i16 + 3] = aw;
    __syncthreads();
    if (tid < DH_) {
        float M = -1e30f;
#pragma unroll
        for (int w = 0; w < 16; ++w) M = fmaxf(M, sm[w]);
        float L = 0.f, X = 0.f;
#pragma unroll
        for (int w = 0; w < 16; ++w) {
            const float ew = __expf(sm[w] - M);
            L += sl[w] * ew;
            X += sacc[w][tid] * ew;
        }
        ctxws[(size_t)b * D_ + h * DH_ + tid] = X / L;
    }
}

// ---------------------------------------------------------------------------
// Kernel C: output projection. out[b][n] = sum_e ctx[b][e]*Wo[e*1024+n] + bo[n]
// grid (4 n-chunks of 256, 4 batch-groups of 8), block 256.
// ---------------------------------------------------------------------------
__global__ __launch_bounds__(256)
void out_proj_kernel(const float* __restrict__ ctx,
                     const float* __restrict__ Wo,
                     const float* __restrict__ bo,
                     float* __restrict__ out)
{
    __shared__ float cs[8 * D_];                       // 32 KB
    const int tid = threadIdx.x;
    const int bg  = blockIdx.y;
    for (int i = tid; i < 8 * D_; i += 256)
        cs[i] = ctx[(bg * 8 + (i >> 10)) * D_ + (i & (D_ - 1))];
    __syncthreads();

    const int n = blockIdx.x * 256 + tid;
    float acc[8];
#pragma unroll
    for (int j = 0; j < 8; ++j) acc[j] = bo[n];
    for (int e = 0; e < D_; ++e) {
        const float w = Wo[e * D_ + n];
#pragma unroll
        for (int j = 0; j < 8; ++j) acc[j] += cs[j * D_ + e] * w;
    }
#pragma unroll
    for (int j = 0; j < 8; ++j)
        out[(size_t)(bg * 8 + j) * D_ + n] = acc[j];
}

// ---------------------------------------------------------------------------
extern "C" void kernel_launch(void* const* d_in, const int* in_sizes, int n_in,
                              void* d_out, int out_size, void* d_ws,
                              size_t ws_size, hipStream_t stream)
{
    (void)in_sizes; (void)n_in; (void)out_size; (void)ws_size;
    const float* x  = (const float*)d_in[0];
    const float* ek = (const float*)d_in[1];
    const float* ev = (const float*)d_in[2];
    const float* Wq = (const float*)d_in[3];
    const float* bq = (const float*)d_in[4];
    const float* Wk = (const float*)d_in[5];
    const float* bk = (const float*)d_in[6];
    const float* Wv = (const float*)d_in[7];
    const float* bv = (const float*)d_in[8];
    const float* Wo = (const float*)d_in[9];
    const float* bo = (const float*)d_in[10];

    float* ws  = (float*)d_ws;
    float* qws = ws;                 // 32*1024
    float* kws = ws + 32768;         // 32*1024
    float* vws = ws + 65536;         // 32*1024
    float* cws = ws + 98304;         // 32*1024

    qkv_proj_kernel<<<dim3(12, 4), 256, 0, stream>>>(x, Wq, bq, Wk, bk, Wv, bv,
                                                     qws, kws, vws);
    decode_attn_kernel<<<dim3(B_ * H_), 256, 0, stream>>>(qws, kws, vws, ek, ev,
                                                          cws);
    out_proj_kernel<<<dim3(4, 4), 256, 0, stream>>>(cws, Wo, bo, (float*)d_out);
}
